// SoftMoEDynamics_17308718203557
// MI455X (gfx1250) — matrix operand order repaired and checked by reading, not compiled, not run on hardware
//
#include <hip/hip_runtime.h>

// ---------------------------------------------------------------------------
// Types for CDNA5 WMMA (wave32) + TDM descriptors
// ---------------------------------------------------------------------------
typedef __attribute__((ext_vector_type(16))) __bf16 bf16x16;
typedef __attribute__((ext_vector_type(8)))  __bf16 bf16x8;
typedef __attribute__((ext_vector_type(8)))  float  f32x8;
typedef __attribute__((ext_vector_type(4)))  unsigned int u32x4;
typedef __attribute__((ext_vector_type(4)))  int    i32x4;
typedef __attribute__((ext_vector_type(8)))  int    i32x8;

#define B_SZ   8192
#define N_EXP  8
#define N_AG   8
#define TOK    320
#define HID    512
#define LAT    256

#if defined(__has_builtin)
#  if __has_builtin(__builtin_amdgcn_tensor_load_to_lds) && \
      __has_builtin(__builtin_amdgcn_s_wait_tensorcnt) && \
      __has_builtin(__builtin_amdgcn_readfirstlane)
#    define HAVE_TDM 1
#  else
#    define HAVE_TDM 0
#  endif
#else
#  define HAVE_TDM 0
#endif

#define WMMA_BF16(C, Am, Bm) \
  C = __builtin_amdgcn_wmma_f32_16x16x32_bf16(false, Am, false, Bm, (short)0, C, false, false)

#if HAVE_TDM
// 2D tile DMA: global (row-major, rows of stride_d0 elements, 2-byte elems)
// -> LDS at byte offset lds_off. tile = tile_d1 rows x tile_d0 elems.
// Descriptor packing per CDNA5 ISA 8.3/8.4 (D# group0/group1), groups 2/3 = 0.
__device__ __forceinline__ void tdm_load_2d_bf16(unsigned lds_off, const void* gaddr,
                                                 unsigned tensor_d0, unsigned tensor_d1,
                                                 unsigned tile_d0, unsigned tile_d1,
                                                 unsigned stride_d0) {
  unsigned long long ga = (unsigned long long)(uintptr_t)gaddr;
  u32x4 g0;
  g0[0] = 1u;                                               // count=1 (valid), user mode
  g0[1] = lds_off;                                          // lds_addr (bytes)
  g0[2] = (unsigned)(ga & 0xffffffffu);                     // global_addr[31:0]
  g0[3] = (unsigned)((ga >> 32) & 0x01ffffffu) | (2u << 30);// global_addr[56:32], type=2
  i32x8 g1;
  g1[0] = (int)(1u << 16);                                  // workgroup_mask=0, data_size=1 (2B)
  g1[1] = (int)((tensor_d0 & 0xffffu) << 16);               // tensor_dim0[15:0] @ bits 63:48
  g1[2] = (int)((tensor_d0 >> 16) | ((tensor_d1 & 0xffffu) << 16)); // dim0 hi | dim1 lo
  g1[3] = (int)((tensor_d1 >> 16) | (tile_d0 << 16));       // dim1 hi | tile_dim0
  g1[4] = (int)(tile_d1 & 0xffffu);                         // tile_dim1, tile_dim2=0
  g1[5] = (int)stride_d0;                                   // tensor_dim0_stride[31:0]
  g1[6] = 0;                                                // stride hi, dim1_stride lo
  g1[7] = 0;
  i32x4 gz = {0, 0, 0, 0};
#if __clang_major__ >= 23
  i32x8 gz8 = {0, 0, 0, 0, 0, 0, 0, 0};
  __builtin_amdgcn_tensor_load_to_lds(g0, g1, gz, gz, gz8, 0);
#else
  __builtin_amdgcn_tensor_load_to_lds(g0, g1, gz, gz, 0);
#endif
}
#endif  // HAVE_TDM

// ---------------------------------------------------------------------------
// Weight cast + transpose:  W (E,K,N) f32  ->  Wt (E,N,K) bf16
// ---------------------------------------------------------------------------
__global__ __launch_bounds__(256)
void tcast_kernel(const float* __restrict__ W, __bf16* __restrict__ Wt,
                  int K, int N) {
  int e   = blockIdx.z;
  int idx = blockIdx.x * 256 + threadIdx.x;
  if (idx >= K * N) return;
  int k = idx / N, n = idx % N;
  Wt[((size_t)e * N + n) * K + k] = (__bf16)W[((size_t)e * K + k) * N + n];
}

// ---------------------------------------------------------------------------
// Routing: tokens -> logits -> dispatch/combine softmaxes -> slot inputs
// ---------------------------------------------------------------------------
__global__ __launch_bounds__(256)
void routing_kernel(const float* __restrict__ latents,   // (B,8,256)
                    const float* __restrict__ actions,   // (B,8,64)
                    const float* __restrict__ phi,       // (320,8)
                    __bf16* __restrict__ slotX,          // (E,B,320)
                    float* __restrict__ comb,            // (B,8,8)
                    int B) {
  __shared__ float tok[N_AG][TOK];
  __shared__ float logit[N_AG][N_EXP];
  __shared__ float disp[N_AG][N_EXP];
  __shared__ float cmb[N_AG][N_EXP];

  int b   = blockIdx.x;
  int tid = threadIdx.x;

  for (int idx = tid; idx < N_AG * TOK; idx += 256) {
    int n = idx / TOK, d = idx % TOK;
    float v = (d < LAT) ? latents[((size_t)b * N_AG + n) * LAT + d]
                        : actions[((size_t)b * N_AG + n) * 64 + (d - LAT)];
    tok[n][d] = v;
  }
  __syncthreads();

  if (tid < 64) {
    int n = tid >> 3, e = tid & 7;
    float s = 0.f;
    for (int d = 0; d < TOK; ++d) s += tok[n][d] * phi[d * N_EXP + e];
    logit[n][e] = s;
  }
  __syncthreads();

  if (tid < 8) {                       // dispatch: softmax over agents n
    int e = tid;
    float mx = -1e30f;
    for (int n = 0; n < N_AG; ++n) mx = fmaxf(mx, logit[n][e]);
    float t[N_AG], s = 0.f;
    for (int n = 0; n < N_AG; ++n) { t[n] = __expf(logit[n][e] - mx); s += t[n]; }
    float inv = 1.f / s;
    for (int n = 0; n < N_AG; ++n) disp[n][e] = t[n] * inv;
  } else if (tid < 16) {               // combine: softmax over experts e
    int n = tid - 8;
    float mx = -1e30f;
    for (int e = 0; e < N_EXP; ++e) mx = fmaxf(mx, logit[n][e]);
    float t[N_EXP], s = 0.f;
    for (int e = 0; e < N_EXP; ++e) { t[e] = __expf(logit[n][e] - mx); s += t[e]; }
    float inv = 1.f / s;
    for (int e = 0; e < N_EXP; ++e) cmb[n][e] = t[e] * inv;
  }
  __syncthreads();

  if (tid < 64) comb[(size_t)b * 64 + tid] = cmb[tid >> 3][tid & 7];

  for (int d = tid; d < TOK; d += 256) {
    float tv[N_AG];
#pragma unroll
    for (int n = 0; n < N_AG; ++n) tv[n] = tok[n][d];
#pragma unroll
    for (int e = 0; e < N_EXP; ++e) {
      float s = 0.f;
#pragma unroll
      for (int n = 0; n < N_AG; ++n) s += disp[n][e] * tv[n];
      slotX[((size_t)e * B + b) * TOK + d] = (__bf16)s;
    }
  }
}

// ---------------------------------------------------------------------------
// Grouped WMMA GEMM, TDM-staged double-buffered B tile:
//   Y[e] = A[e] (M,K) x Wt[e]^T   (Wt stored (E,N,K))
// block = 8 waves, tile 256(M) x 64(N); wave: 32 rows x 4 col-tiles = 8 accums.
// Pipeline per 64-K chunk: issue TDM for chunk c+1 into the other LDS buffer,
// s_wait_tensorcnt 1 (drains chunk c only), barrier, 16x WMMA from LDS.
// ---------------------------------------------------------------------------
__global__ __launch_bounds__(256)
void wmma_gemm_kernel(const __bf16* __restrict__ A,   // (E,M,K)
                      const __bf16* __restrict__ Wt,  // (E,N,K)
                      void* __restrict__ Y,           // (E,M,N) bf16 or f32
                      int M, int K, int N, int outF32) {
  __shared__ __bf16 sB[2][64 * 64];   // double-buffered B tile (2 x 8 KB)

  const int e    = blockIdx.z;
  const int lane = threadIdx.x & 31;
  const int wv   = threadIdx.x >> 5;
  const int l15  = lane & 15;
  const int hsel = lane >> 4;                  // 0: lanes 0-15, 1: lanes 16-31

  const int rowBase = blockIdx.x * 256 + wv * 32;
  const int colBase = blockIdx.y * 64;

  const __bf16* Ae = A  + (size_t)e * M * K;
  const __bf16* We = Wt + (size_t)e * N * K;

  // A: lanes 0-15 hold K {0..15}; lanes 16-31 hold K {16..31}
  const __bf16* aPtr0 = Ae + (size_t)(rowBase + l15) * K + hsel * 16;
  const __bf16* aPtr1 = Ae + (size_t)(rowBase + 16 + l15) * K + hsel * 16;

#if HAVE_TDM
  const int wvs = __builtin_amdgcn_readfirstlane(wv);      // wave-uniform scalar
  const unsigned ldsB0 = (unsigned)(uintptr_t)(&sB[0][0]);
  const unsigned ldsB1 = (unsigned)(uintptr_t)(&sB[1][0]);
#endif

  f32x8 c0a = {}, c1a = {}, c2a = {}, c3a = {};
  f32x8 c0b = {}, c1b = {}, c2b = {}, c3b = {};

  const int NC = K >> 6;               // 64-wide K chunks (K is 320 or 512)

  // ---- prologue: stage chunk 0 into buffer 0 ----
#if HAVE_TDM
  tdm_load_2d_bf16(ldsB0 + (unsigned)(wvs * 8 * 64 * 2),
                   We + (size_t)(colBase + wvs * 8) * K,
                   (unsigned)K, 8u, 64u, 8u, (unsigned)K);
#else
  for (int i = threadIdx.x; i < 64 * 8; i += 256) {
    int rr = i >> 3, cc = (i & 7) * 8;
    *(bf16x8*)&sB[0][rr * 64 + cc] =
        *(const bf16x8*)(We + (size_t)(colBase + rr) * K + cc);
  }
#endif

  for (int c = 0; c < NC; ++c) {
    const int k0  = c << 6;
    const int cur = c & 1;

    // ---- issue next chunk into the other buffer, then drain current ----
#if HAVE_TDM
    if (c + 1 < NC) {
      tdm_load_2d_bf16((cur ? ldsB0 : ldsB1) + (unsigned)(wvs * 8 * 64 * 2),
                       We + (size_t)(colBase + wvs * 8) * K + k0 + 64,
                       (unsigned)K, 8u, 64u, 8u, (unsigned)K);
      __builtin_amdgcn_s_wait_tensorcnt(1);   // chunk c done; c+1 still in flight
    } else {
      __builtin_amdgcn_s_wait_tensorcnt(0);
    }
#else
    if (c + 1 < NC) {
      for (int i = threadIdx.x; i < 64 * 8; i += 256) {
        int rr = i >> 3, cc = (i & 7) * 8;
        *(bf16x8*)&sB[cur ^ 1][rr * 64 + cc] =
            *(const bf16x8*)(We + (size_t)(colBase + rr) * K + k0 + 64 + cc);
      }
    }
#endif
    __syncthreads();   // chunk c published in sB[cur]

    // prefetch next A chunk (speculative; dropped if OOB)
    __builtin_prefetch(aPtr0 + k0 + 64, 0, 1);
    __builtin_prefetch(aPtr1 + k0 + 64, 0, 1);

    const __bf16* sBc = &sB[cur][0];
#pragma unroll
    for (int kk = 0; kk < 64; kk += 32) {
      bf16x16 a0, a1;
      {
        bf16x8 lo = *(const bf16x8*)(aPtr0 + k0 + kk);
        bf16x8 hi = *(const bf16x8*)(aPtr0 + k0 + kk + 8);
#pragma unroll
        for (int i = 0; i < 8; ++i) { a0[i] = lo[i]; a0[i + 8] = hi[i]; }
      }
      {
        bf16x8 lo = *(const bf16x8*)(aPtr1 + k0 + kk);
        bf16x8 hi = *(const bf16x8*)(aPtr1 + k0 + kk + 8);
#pragma unroll
        for (int i = 0; i < 8; ++i) { a1[i] = lo[i]; a1[i + 8] = hi[i]; }
      }

      // B: lanes 0-15 K kk..kk+15, lanes 16-31 K kk+16..kk+31 (contig LDS row)
      bf16x16 b0 = *(const bf16x16*)&sBc[( 0 + l15) * 64 + hsel * 16 + kk];
      bf16x16 b1 = *(const bf16x16*)&sBc[(16 + l15) * 64 + hsel * 16 + kk];
      bf16x16 b2 = *(const bf16x16*)&sBc[(32 + l15) * 64 + hsel * 16 + kk];
      bf16x16 b3 = *(const bf16x16*)&sBc[(48 + l15) * 64 + hsel * 16 + kk];

      WMMA_BF16(c0a, a0, b0); WMMA_BF16(c0b, a1, b0);
      WMMA_BF16(c1a, a0, b1); WMMA_BF16(c1b, a1, b1);
      WMMA_BF16(c2a, a0, b2); WMMA_BF16(c2b, a1, b2);
      WMMA_BF16(c3a, a0, b3); WMMA_BF16(c3b, a1, b3);
    }
    __syncthreads();   // readers done with sB[cur] before it is re-filled
  }

  // C/D layout: lane n (0-15) col = colBase+t*16+n, VGPR r = row r (+8 upper half)
  auto storeTile = [&](const f32x8& cv, int t, int rbase) {
    int col = colBase + t * 16 + l15;
#pragma unroll
    for (int r = 0; r < 8; ++r) {
      int row = rbase + r + hsel * 8;
      size_t idx = ((size_t)e * M + row) * N + col;
      if (outF32) ((float*)Y)[idx] = cv[r];
      else        ((__bf16*)Y)[idx] = (__bf16)cv[r];
    }
  };
  storeTile(c0a, 0, rowBase);      storeTile(c1a, 1, rowBase);
  storeTile(c2a, 2, rowBase);      storeTile(c3a, 3, rowBase);
  storeTile(c0b, 0, rowBase + 16); storeTile(c1b, 1, rowBase + 16);
  storeTile(c2b, 2, rowBase + 16); storeTile(c3b, 3, rowBase + 16);
}

// ---------------------------------------------------------------------------
// In-place bias + LayerNorm + mish over rows of 512 (one wave per row).
// ---------------------------------------------------------------------------
__global__ __launch_bounds__(256)
void ln_mish_kernel(__bf16* __restrict__ X,          // (E*M, 512) in-place
                    const float* __restrict__ bias,  // (E,512)
                    const float* __restrict__ gamma, // (E,512)
                    const float* __restrict__ beta,  // (E,512)
                    int Mper) {
  int lane = threadIdx.x & 31;
  int wv   = threadIdx.x >> 5;
  int row  = blockIdx.x * 8 + wv;
  int e    = row / Mper;

  __bf16* xr = X + (size_t)row * HID;
  const float* bp  = bias  + (size_t)e * HID;
  const float* gp  = gamma + (size_t)e * HID;
  const float* bep = beta  + (size_t)e * HID;

  float v[16], s = 0.f, sq = 0.f;
#pragma unroll
  for (int i = 0; i < 16; ++i) {
    int col = i * 32 + lane;
    float t = (float)xr[col] + bp[col];
    v[i] = t; s += t; sq += t * t;
  }
#pragma unroll
  for (int off = 16; off; off >>= 1) {
    s  += __shfl_xor(s, off);
    sq += __shfl_xor(sq, off);
  }
  float mean = s * (1.0f / 512.0f);
  float var  = sq * (1.0f / 512.0f) - mean * mean;
  float rstd = rsqrtf(var + 1e-5f);

#pragma unroll
  for (int i = 0; i < 16; ++i) {
    int col = i * 32 + lane;
    float y  = (v[i] - mean) * rstd * gp[col] + bep[col];
    float sp = (y > 20.f) ? y : log1pf(__expf(y));   // softplus
    xr[col] = (__bf16)(y * tanhf(sp));               // mish
  }
}

// ---------------------------------------------------------------------------
// bias + simnorm (softmax over contiguous groups of 8); rearrange to (B,E,256)
// ---------------------------------------------------------------------------
__global__ __launch_bounds__(256)
void simnorm_kernel(const float* __restrict__ Y,   // (E,M,256)
                    const float* __restrict__ b3,  // (E,256)
                    float* __restrict__ S,         // (B,E,256)
                    int Mper) {
  int lane = threadIdx.x & 31;
  int wv   = threadIdx.x >> 5;
  int row  = blockIdx.x * 8 + wv;         // row in [0, E*M)
  int e = row / Mper, m = row % Mper;

  const float* yr = Y  + (size_t)row * LAT + lane * 8;
  const float* bp = b3 + (size_t)e  * LAT + lane * 8;

  float g[8], mx = -1e30f;
#pragma unroll
  for (int j = 0; j < 8; ++j) { g[j] = yr[j] + bp[j]; mx = fmaxf(mx, g[j]); }
  float ss = 0.f;
#pragma unroll
  for (int j = 0; j < 8; ++j) { g[j] = __expf(g[j] - mx); ss += g[j]; }
  float inv = 1.0f / ss;

  float* so = S + ((size_t)m * N_EXP + e) * LAT + lane * 8;
#pragma unroll
  for (int j = 0; j < 8; ++j) so[j] = g[j] * inv;
}

// ---------------------------------------------------------------------------
// Combine: out[b,n,:] = sum_e comb[b,n,e] * S[b,e,:]
// ---------------------------------------------------------------------------
__global__ __launch_bounds__(256)
void combine_kernel(const float* __restrict__ comb, // (B,8,8)
                    const float* __restrict__ S,    // (B,8,256)
                    float* __restrict__ out,        // (B,8,256)
                    int B) {
  __shared__ float sS[N_EXP][LAT];
  __shared__ float sC[N_AG][N_EXP];
  int b = blockIdx.x, tid = threadIdx.x;

  for (int idx = tid; idx < N_EXP * LAT; idx += 256)
    sS[idx >> 8][idx & 255] = S[(size_t)b * (N_EXP * LAT) + idx];
  if (tid < 64) sC[tid >> 3][tid & 7] = comb[(size_t)b * 64 + tid];
  __syncthreads();

  int d = tid;  // 0..255
#pragma unroll
  for (int n = 0; n < N_AG; ++n) {
    float acc = 0.f;
#pragma unroll
    for (int e = 0; e < N_EXP; ++e) acc += sC[n][e] * sS[e][d];
    out[((size_t)b * N_AG + n) * LAT + d] = acc;
  }
}

// ---------------------------------------------------------------------------
// Host launch
// ---------------------------------------------------------------------------
extern "C" void kernel_launch(void* const* d_in, const int* in_sizes, int n_in,
                              void* d_out, int out_size, void* d_ws, size_t ws_size,
                              hipStream_t stream) {
  const float* latents = (const float*)d_in[0];
  const float* actions = (const float*)d_in[1];
  const float* phi     = (const float*)d_in[2];
  const float* W1 = (const float*)d_in[3];
  const float* b1 = (const float*)d_in[4];
  const float* g1 = (const float*)d_in[5];
  const float* be1 = (const float*)d_in[6];
  const float* W2 = (const float*)d_in[7];
  const float* b2 = (const float*)d_in[8];
  const float* g2 = (const float*)d_in[9];
  const float* be2 = (const float*)d_in[10];
  const float* W3 = (const float*)d_in[11];
  const float* b3 = (const float*)d_in[12];
  float* out = (float*)d_out;

  char* ws = (char*)d_ws;
  size_t o = 0;
  __bf16* slotX = (__bf16*)(ws + o); o += (size_t)N_EXP * B_SZ * TOK * 2;   // 41.9 MB
  float*  comb  = (float*)(ws + o);  o += (size_t)B_SZ * 64 * 4;            //  2.1 MB
  __bf16* W1t   = (__bf16*)(ws + o); o += (size_t)N_EXP * HID * TOK * 2;    //  2.6 MB
  __bf16* W2t   = (__bf16*)(ws + o); o += (size_t)N_EXP * HID * HID * 2;    //  4.2 MB
  __bf16* W3t   = (__bf16*)(ws + o); o += (size_t)N_EXP * LAT * HID * 2;    //  2.1 MB
  char*   X0    = ws + o;            o += (size_t)N_EXP * B_SZ * HID * 2;   // 67.1 MB
  char*   X1    = ws + o;            o += (size_t)N_EXP * B_SZ * HID * 2;   // 67.1 MB

  // Weight transpose+cast (rerun each call; no caching allowed)
  tcast_kernel<<<dim3((TOK * HID + 255) / 256, 1, N_EXP), 256, 0, stream>>>(W1, W1t, TOK, HID);
  tcast_kernel<<<dim3((HID * HID + 255) / 256, 1, N_EXP), 256, 0, stream>>>(W2, W2t, HID, HID);
  tcast_kernel<<<dim3((HID * LAT + 255) / 256, 1, N_EXP), 256, 0, stream>>>(W3, W3t, HID, LAT);

  // Routing + dispatch mixing
  routing_kernel<<<B_SZ, 256, 0, stream>>>(latents, actions, phi, slotX, comb, B_SZ);

  // Layer 1: (E,8192,320) x (320,512) -> bf16 X0, then LN+mish in place
  wmma_gemm_kernel<<<dim3(B_SZ / 256, HID / 64, N_EXP), 256, 0, stream>>>(
      slotX, W1t, X0, B_SZ, TOK, HID, 0);
  ln_mish_kernel<<<(N_EXP * B_SZ) / 8, 256, 0, stream>>>((__bf16*)X0, b1, g1, be1, B_SZ);

  // Layer 2: (E,8192,512) x (512,512) -> bf16 X1, then LN+mish in place
  wmma_gemm_kernel<<<dim3(B_SZ / 256, HID / 64, N_EXP), 256, 0, stream>>>(
      (__bf16*)X0, W2t, X1, B_SZ, HID, HID, 0);
  ln_mish_kernel<<<(N_EXP * B_SZ) / 8, 256, 0, stream>>>((__bf16*)X1, b2, g2, be2, B_SZ);

  // Layer 3: (E,8192,512) x (512,256) -> f32 into X0, then bias+simnorm -> S in X1
  wmma_gemm_kernel<<<dim3(B_SZ / 256, LAT / 64, N_EXP), 256, 0, stream>>>(
      (__bf16*)X1, W3t, X0, B_SZ, HID, LAT, 1);
  simnorm_kernel<<<(N_EXP * B_SZ) / 8, 256, 0, stream>>>((const float*)X0, b3, (float*)X1, B_SZ);

  // Combine slots back to agent latents
  combine_kernel<<<B_SZ, 256, 0, stream>>>(comb, (const float*)X1, out, B_SZ);
}
